// HMM_ensemble_56831007261219
// MI455X (gfx1250) — matrix operand order repaired and checked by reading, not compiled, not run on hardware
//
#include <hip/hip_runtime.h>
#include <hip/hip_bf16.h>
#include <math.h>
#include <float.h>

typedef __attribute__((ext_vector_type(16))) _Float16 v16h;
typedef __attribute__((ext_vector_type(8)))  float    v8f;
typedef __attribute__((ext_vector_type(2)))  __fp16   fp16x2;

#define LDIM 4
#define NST  64
#define MDIM 256
#define TLEN 1024

#define LOG2E 1.4426950408889634f
#define LN2   0.6931471805599453f

__device__ __forceinline__ unsigned pack_h2(float a, float b) {
  fp16x2 p = __builtin_amdgcn_cvt_pkrtz(a, b);   // v_cvt_pk_f16_f32
  return __builtin_bit_cast(unsigned, p);
}

__device__ __forceinline__ float max3f(float a, float b, float c) {
  return fmaxf(fmaxf(a, b), c);                  // matches v_max3_num_f32
}

// One wave per block. Block b = (l, cb): latent l, columns m = cb*16 .. cb*16+15.
// beta = alpha * log2(e) kept in registers in WMMA C/D layout:
// lane (n = lane&15, h = lane>>4), j = it*8 + r  <->  state i = it*16 + 8h + r.
__global__ __launch_bounds__(32) void hmm_fwd_kernel(
    const int*   __restrict__ x,
    const float* __restrict__ lambdas,
    const float* __restrict__ log_transition,
    const float* __restrict__ log_state_priors,
    float*       __restrict__ wsLog)
{
  const int lane = threadIdx.x;
  const int n = lane & 15;
  const int h = lane >> 4;
  const int l  = blockIdx.x >> 4;
  const int cb = blockIdx.x & 15;
  const int m  = cb * 16 + n;

  // 64 x 16 f16 staging buffer for exp2(beta - c), stored in B-operand order.
  __shared__ __align__(32) unsigned short Ebuf[NST * 16];

  // ---- Preload exp(log_transition[l]) as f16 WMMA A-fragments (constant over t).
  // A 16x32 f16 layout: lane row = n; element e -> K = 16h + e (+32/chunk).
  v16h afrag[4][2];
  #pragma unroll
  for (int it = 0; it < 4; ++it) {
    #pragma unroll
    for (int ch = 0; ch < 2; ++ch) {
      #pragma unroll
      for (int e = 0; e < 16; ++e) {
        int K = ch * 32 + 16 * h + e;
        int i = it * 16 + n;
        float v = __builtin_amdgcn_exp2f(
            log_transition[l * NST * NST + i * NST + K] * LOG2E);
        afrag[it][ch][e] = (_Float16)v;
      }
    }
  }

  // ---- Per-lane Poisson constants in log2 units (lgamma folded in at finalize).
  float llg2[32], lam2[32], beta[32];
  #pragma unroll
  for (int it = 0; it < 4; ++it) {
    #pragma unroll
    for (int r = 0; r < 8; ++r) {
      int i = it * 16 + 8 * h + r;
      float lv = lambdas[l * NST * MDIM + i * MDIM + m];
      lam2[it * 8 + r] = lv * LOG2E;                    // lambda * log2(e)
      llg2[it * 8 + r] = __builtin_amdgcn_logf(lv);     // log2(lambda)
    }
  }

  // ---- t = 0: beta0 = log2e * (em0 + log_state_priors)
  {
    float xf = (float)x[m];
    #pragma unroll
    for (int it = 0; it < 4; ++it) {
      #pragma unroll
      for (int r = 0; r < 8; ++r) {
        int i = it * 16 + 8 * h + r;
        beta[it * 8 + r] = xf * llg2[it * 8 + r] - lam2[it * 8 + r]
                         + log_state_priors[l * NST + i] * LOG2E;
      }
    }
  }

  v8f zeroacc = {};
  float xf = (float)x[1 * MDIM + m];   // prefetched operand for t = 1

  #pragma unroll 1
  for (int t = 1; t < TLEN; ++t) {
    // column max c over all 64 states: 3-way max tree (v_max3) + cross-half shuffle
    float p0 = max3f(beta[0],  beta[1],  beta[2]);
    float p1 = max3f(beta[3],  beta[4],  beta[5]);
    float p2 = max3f(beta[6],  beta[7],  beta[8]);
    float p3 = max3f(beta[9],  beta[10], beta[11]);
    float p4 = max3f(beta[12], beta[13], beta[14]);
    float p5 = max3f(beta[15], beta[16], beta[17]);
    float p6 = max3f(beta[18], beta[19], beta[20]);
    float p7 = max3f(beta[21], beta[22], beta[23]);
    float p8 = max3f(beta[24], beta[25], beta[26]);
    float p9 = max3f(beta[27], beta[28], beta[29]);
    float q0 = max3f(p0, p1, p2);
    float q1 = max3f(p3, p4, p5);
    float q2 = max3f(p6, p7, p8);
    float q3 = max3f(p9, beta[30], beta[31]);
    float mxa = fmaxf(max3f(q0, q1, q2), q3);
    float c = fmaxf(mxa, __shfl_xor(mxa, 16, 32));

    // E = exp2(beta - c) as f16 -> LDS in B-operand order.
    // Lane's 8 consecutive k (= it*16 + 8h + r, r=0..7) are contiguous halves.
    #pragma unroll
    for (int it = 0; it < 4; ++it) {
      unsigned w0 = pack_h2(__builtin_amdgcn_exp2f(beta[it*8+0]-c),
                            __builtin_amdgcn_exp2f(beta[it*8+1]-c));
      unsigned w1 = pack_h2(__builtin_amdgcn_exp2f(beta[it*8+2]-c),
                            __builtin_amdgcn_exp2f(beta[it*8+3]-c));
      unsigned w2 = pack_h2(__builtin_amdgcn_exp2f(beta[it*8+4]-c),
                            __builtin_amdgcn_exp2f(beta[it*8+5]-c));
      unsigned w3 = pack_h2(__builtin_amdgcn_exp2f(beta[it*8+6]-c),
                            __builtin_amdgcn_exp2f(beta[it*8+7]-c));
      uint4* dst = (uint4*)&Ebuf[it * 256 + n * 16 + h * 8];
      *dst = make_uint4(w0, w1, w2, w3);
    }
    __syncthreads();

    // Prefetch next x: issued here so next iteration's barrier-wait finds it done.
    const int tn = (t + 1 < TLEN) ? (t + 1) : (TLEN - 1);
    const int xnext = x[tn * MDIM + m];

    // B-operand loads: lane (col n, half h) reads 16 contiguous halves per K-chunk.
    const v16h b0 = *(const v16h*)&Ebuf[0 * 512 + h * 256 + n * 16];
    const v16h b1 = *(const v16h*)&Ebuf[1 * 512 + h * 256 + n * 16];

    #pragma unroll
    for (int it = 0; it < 4; ++it) {
      v8f acc = __builtin_amdgcn_wmma_f32_16x16x32_f16(
                  false, afrag[it][0], false, b0, (short)0, zeroacc, false, false);
      acc = __builtin_amdgcn_wmma_f32_16x16x32_f16(
                  false, afrag[it][1], false, b1, (short)0, acc, false, false);
      #pragma unroll
      for (int r = 0; r < 8; ++r) {
        float d = __builtin_amdgcn_fmed3f(acc[r], 1e-30f, FLT_MAX); // 1-op clamp
        beta[it*8+r] = xf * llg2[it*8+r] - lam2[it*8+r]
                     + __builtin_amdgcn_logf(d) + c;
      }
    }
    __syncthreads();   // WAR fence: next iteration's stores vs this iteration's loads
    xf = (float)xnext;
  }

  // logpx[l,m] = ln2 * logsumexp2 over the 64 states
  float mxa = beta[0];
  #pragma unroll
  for (int j = 1; j < 32; ++j) mxa = fmaxf(mxa, beta[j]);
  mxa = fmaxf(mxa, __shfl_xor(mxa, 16, 32));
  float s = 0.f;
  #pragma unroll
  for (int j = 0; j < 32; ++j) s += __builtin_amdgcn_exp2f(beta[j] - mxa);
  s += __shfl_xor(s, 16, 32);
  if (lane < 16)
    wsLog[l * MDIM + m] = LN2 * (mxa + __builtin_amdgcn_logf(s));
}

// Finalize: fold in the lgamma(x+1) sums (constant over states, counted once per
// latent), add log ensemble probabilities, logsumexp over m -> scalar.
__global__ __launch_bounds__(256) void hmm_finalize_kernel(
    const int*   __restrict__ x,
    const float* __restrict__ eprob,   // (M, L)
    const float* __restrict__ wsLog,   // (L, M)
    float*       __restrict__ out)
{
  const int m = threadIdx.x;
  float S = 0.f;
  for (int t = 0; t < TLEN; ++t)
    S += lgammaf((float)x[t * MDIM + m] + 1.0f);

  float tot = 0.f;
  #pragma unroll
  for (int l = 0; l < LDIM; ++l)
    tot += logf(eprob[m * LDIM + l] + 1e-32f) + (wsLog[l * MDIM + m] - S);

  __shared__ float sm[MDIM];
  sm[m] = tot;
  __syncthreads();
  for (int s2 = 128; s2 > 0; s2 >>= 1) {
    if (m < s2) sm[m] = fmaxf(sm[m], sm[m + s2]);
    __syncthreads();
  }
  float mx = sm[0];
  __syncthreads();
  sm[m] = expf(tot - mx);
  __syncthreads();
  for (int s2 = 128; s2 > 0; s2 >>= 1) {
    if (m < s2) sm[m] += sm[m + s2];
    __syncthreads();
  }
  if (m == 0) out[0] = mx + logf(sm[0]);
}

extern "C" void kernel_launch(void* const* d_in, const int* in_sizes, int n_in,
                              void* d_out, int out_size, void* d_ws, size_t ws_size,
                              hipStream_t stream) {
  (void)in_sizes; (void)n_in; (void)out_size; (void)ws_size;
  const int*   x       = (const int*)  d_in[0];
  const float* lambdas = (const float*)d_in[1];
  const float* log_tr  = (const float*)d_in[2];
  const float* log_pri = (const float*)d_in[3];
  const float* eprob   = (const float*)d_in[4];
  float* wsLog = (float*)d_ws;                 // L*M floats = 4 KB scratch
  float* out   = (float*)d_out;

  hmm_fwd_kernel<<<dim3(LDIM * (MDIM / 16)), dim3(32), 0, stream>>>(
      x, lambdas, log_tr, log_pri, wsLog);
  hmm_finalize_kernel<<<dim3(1), dim3(MDIM), 0, stream>>>(x, eprob, wsLog, out);
}
